// MultiHeadAttention_26482768347222
// MI455X (gfx1250) — compile-verified, matrix operand order checked by reading, not run on hardware
//
#include <hip/hip_runtime.h>

typedef _Float16 v16h __attribute__((ext_vector_type(16)));
typedef float    v8f  __attribute__((ext_vector_type(8)));
typedef _Float16 h4   __attribute__((ext_vector_type(4)));
typedef int      v4i  __attribute__((ext_vector_type(4)));

#define BATCH   2
#define SEQ     2048
#define DIM     1024
#define HEADS   16
#define DHEAD   64
#define INNER   1024
#define QKVN    3072
#define FRAME   256
#define ROWS    (BATCH*SEQ)   /* 4096 */
#define LN_EPS  1e-5f

#if __has_builtin(__builtin_amdgcn_global_load_async_to_lds_b128)
#define HAVE_ASYNC_LDS 1
#else
#define HAVE_ASYNC_LDS 0
#endif

union FragU { v16h h; uint4 q[2]; };

// Fragment load from a pre-offset row pointer (base already includes
// row*ld + 8*(lane>>4)); k0 selects the 32-wide k-chunk. Two b128 loads.
__device__ inline v16h load_frag_p(const _Float16* p, int k0) {
  FragU f;
  f.q[0] = *(const uint4*)(p + k0);
  f.q[1] = *(const uint4*)(p + k0 + 16);
  return f.h;
}

// Generic fragment load (used for small LDS tiles in attention).
__device__ inline v16h load_frag16(const _Float16* __restrict__ base, int row,
                                   int ld, int k0, int lane) {
  return load_frag_p(base + (size_t)row * ld + 8 * (lane >> 4), k0);
}

__device__ inline v8f wmma_f16(v16h a, v16h b, v8f c) {
  return __builtin_amdgcn_wmma_f32_16x16x32_f16(false, a, false, b, (short)0, c,
                                                false, false);
}

// Cooperative 16-byte global->LDS copy element (async on CDNA5 if available).
__device__ inline void copy16_to_lds(const _Float16* gsrc, _Float16* ldst) {
#if HAVE_ASYNC_LDS
  __builtin_amdgcn_global_load_async_to_lds_b128(
      (__attribute__((address_space(1))) v4i*)gsrc,
      (__attribute__((address_space(3))) v4i*)ldst, 0, 0);
#else
  *(uint4*)ldst = *(const uint4*)gsrc;
#endif
}

__device__ inline void wait_async_copies() {
#if HAVE_ASYNC_LDS
#if __has_builtin(__builtin_amdgcn_s_wait_asynccnt)
  __builtin_amdgcn_s_wait_asynccnt(0);
#else
  asm volatile("s_wait_asynccnt 0x0" ::: "memory");
#endif
#endif
}

// ---------------- Kernel 1: LayerNorm -> f16 ----------------
__global__ void ln_kernel(const float* __restrict__ x, const float* __restrict__ g,
                          const float* __restrict__ bb, _Float16* __restrict__ xn) {
  __shared__ float s1[256];
  __shared__ float s2[256];
  int row = blockIdx.x;
  int tid = threadIdx.x;
  const float4* xr = (const float4*)(x + (size_t)row * DIM);
  float4 v = xr[tid];
  s1[tid] = v.x + v.y + v.z + v.w;
  s2[tid] = v.x * v.x + v.y * v.y + v.z * v.z + v.w * v.w;
  __syncthreads();
  for (int off = 128; off > 0; off >>= 1) {
    if (tid < off) { s1[tid] += s1[tid + off]; s2[tid] += s2[tid + off]; }
    __syncthreads();
  }
  float mean = s1[0] * (1.0f / DIM);
  float var  = s2[0] * (1.0f / DIM) - mean * mean;
  float inv  = rsqrtf(var + LN_EPS);
  float4 gv = ((const float4*)g)[tid];
  float4 bv = ((const float4*)bb)[tid];
  h4 o;
  o[0] = (_Float16)((v.x - mean) * inv * gv.x + bv.x);
  o[1] = (_Float16)((v.y - mean) * inv * gv.y + bv.y);
  o[2] = (_Float16)((v.z - mean) * inv * gv.z + bv.z);
  o[3] = (_Float16)((v.w - mean) * inv * gv.w + bv.w);
  ((h4*)(xn + (size_t)row * DIM))[tid] = o;
}

// ---------------- Kernel 2: transpose f32 [K,N] -> f16 [N,K] ----------------
__global__ void wtrans_kernel(const float* __restrict__ src, _Float16* __restrict__ dst,
                              int K, int N) {
  size_t idx = (size_t)blockIdx.x * 256 + threadIdx.x;
  if (idx >= (size_t)K * N) return;
  int n = (int)(idx / K);
  int k = (int)(idx % K);
  dst[idx] = (_Float16)src[(size_t)k * N + n];
}

// ------- Kernel 3: QKV GEMM, 64x32 register-blocked per wave (8 WMMA/step) ---
__global__ void qkv_gemm(const _Float16* __restrict__ xn, const _Float16* __restrict__ wt,
                         _Float16* __restrict__ qb, _Float16* __restrict__ kb,
                         _Float16* __restrict__ vtb) {
  int lane = threadIdx.x & 31, wave = threadIdx.x >> 5;
  int t  = blockIdx.x * 4 + wave;
  int mt = t / (QKVN / 32);
  int nt = t % (QKVN / 32);
  int m0 = mt * 64, n0 = nt * 32;
  int r16 = lane & 15, half = lane >> 4;

  // Hoisted per-fragment row pointers (k addressed via immediate offsets).
  const _Float16* a0 = xn + (size_t)(m0      + r16) * DIM + 8 * half;
  const _Float16* a1 = a0 + 16 * DIM;
  const _Float16* a2 = a0 + 32 * DIM;
  const _Float16* a3 = a0 + 48 * DIM;
  const _Float16* b0 = wt + (size_t)(n0 + r16) * DIM + 8 * half;
  const _Float16* b1 = b0 + 16 * DIM;

  v8f acc[4][2] = {};
  for (int k0 = 0; k0 < DIM; k0 += 32) {
    v16h A0 = load_frag_p(a0, k0);
    v16h A1 = load_frag_p(a1, k0);
    v16h A2 = load_frag_p(a2, k0);
    v16h A3 = load_frag_p(a3, k0);
    v16h B0 = load_frag_p(b0, k0);
    acc[0][0] = wmma_f16(A0, B0, acc[0][0]);
    acc[1][0] = wmma_f16(A1, B0, acc[1][0]);
    acc[2][0] = wmma_f16(A2, B0, acc[2][0]);
    acc[3][0] = wmma_f16(A3, B0, acc[3][0]);
    v16h B1 = load_frag_p(b1, k0);
    acc[0][1] = wmma_f16(A0, B1, acc[0][1]);
    acc[1][1] = wmma_f16(A1, B1, acc[1][1]);
    acc[2][1] = wmma_f16(A2, B1, acc[2][1]);
    acc[3][1] = wmma_f16(A3, B1, acc[3][1]);
  }

#pragma unroll
  for (int j = 0; j < 2; ++j) {
    int col = n0 + 16 * j + r16;
    int seg = col / INNER;            // uniform per 16-col tile
    int n2  = col % INNER;
    int h = n2 / DHEAD, d = n2 % DHEAD;
#pragma unroll
    for (int i = 0; i < 4; ++i) {
#pragma unroll
      for (int r = 0; r < 8; ++r) {
        int row = m0 + 16 * i + r + 8 * half;
        int bi = row / SEQ, pos = row % SEQ;
        float val = acc[i][j][r];
        if (seg == 0) {
          qb[((size_t)(bi * HEADS + h) * SEQ + pos) * DHEAD + d] = (_Float16)(val * 0.125f);
        } else if (seg == 1) {
          kb[((size_t)(bi * HEADS + h) * SEQ + pos) * DHEAD + d] = (_Float16)val;
        } else {
          vtb[((size_t)(bi * HEADS + h) * DHEAD + d) * SEQ + pos] = (_Float16)val;
        }
      }
    }
  }
}

// ------- Kernel 4: block-cooperative block-causal flash attention -----------
// Block = 4 waves, 64 consecutive q rows of one (b,h). K/V tiles staged in LDS
// once per block (async global->LDS when available).
__global__ void attn_kernel(const _Float16* __restrict__ qb, const _Float16* __restrict__ kb,
                            const _Float16* __restrict__ vtb, _Float16* __restrict__ ob) {
  __shared__ __align__(16) _Float16 sK[32 * 64];    // keys x dhead
  __shared__ __align__(16) _Float16 sV[64 * 32];    // dhead x keys (V^T)
  __shared__ __align__(16) _Float16 pLds[4][16 * 32];

  int lane = threadIdx.x & 31, wave = threadIdx.x >> 5;
  int tid  = threadIdx.x;
  int qblk = blockIdx.x & (SEQ / 64 - 1);           // 32 blocks of 64 q rows
  int h    = (blockIdx.x >> 5) & (HEADS - 1);
  int b    = blockIdx.x >> 9;
  int q0   = qblk * 64 + wave * 16;
  int kmax = ((qblk * 64) >> 8) * FRAME + FRAME;    // same for all 4 waves

  const _Float16* qbase = qb  + (size_t)(b * HEADS + h) * SEQ * DHEAD;
  const _Float16* kbase = kb  + (size_t)(b * HEADS + h) * SEQ * DHEAD;
  const _Float16* vbase = vtb + (size_t)(b * HEADS + h) * DHEAD * SEQ;
  int r16 = lane & 15, half = lane >> 4;

  v16h aq0 = load_frag16(qbase, q0 + r16, DHEAD, 0, lane);
  v16h aq1 = load_frag16(qbase, q0 + r16, DHEAD, 32, lane);

  float mrow[8], lrow[8];
  v8f o0 = {}, o1 = {}, o2 = {}, o3 = {};
  for (int r = 0; r < 8; ++r) { mrow[r] = -1e30f; lrow[r] = 0.0f; }

  for (int kk = 0; kk < kmax; kk += 32) {
    // ---- cooperative K/V tile fill: 32x64 K + 64x32 V^T (8KB), 4x b128/thread
#pragma unroll
    for (int u = 0; u < 2; ++u) {
      int q4 = tid + u * 128;                 // uint4 index 0..255
      int krow = q4 >> 3, kcol = (q4 & 7) * 8;
      copy16_to_lds(kbase + (size_t)(kk + krow) * DHEAD + kcol, sK + krow * 64 + kcol);
      int vrow = q4 >> 2, vcol = (q4 & 3) * 8;
      copy16_to_lds(vbase + (size_t)vrow * SEQ + kk + vcol, sV + vrow * 32 + vcol);
    }
    wait_async_copies();
    __syncthreads();

    // ---- S = Q K^T (fragments from LDS)
    v16h bk00 = load_frag16(sK, r16,      64, 0,  lane);
    v16h bk01 = load_frag16(sK, r16,      64, 32, lane);
    v16h bk10 = load_frag16(sK, 16 + r16, 64, 0,  lane);
    v16h bk11 = load_frag16(sK, 16 + r16, 64, 32, lane);
    v8f s0 = {}, s1 = {};
    s0 = wmma_f16(aq0, bk00, s0);
    s0 = wmma_f16(aq1, bk01, s0);
    s1 = wmma_f16(aq0, bk10, s1);
    s1 = wmma_f16(aq1, bk11, s1);

    // ---- online softmax (row stats via shfl within 16-lane half-groups)
#pragma unroll
    for (int r = 0; r < 8; ++r) {
      float v = fmaxf(s0[r], s1[r]);
      for (int m = 1; m < 16; m <<= 1) v = fmaxf(v, __shfl_xor(v, m, 32));
      float mnew  = fmaxf(mrow[r], v);
      float alpha = __expf(mrow[r] - mnew);
      float p0 = __expf(s0[r] - mnew);
      float p1 = __expf(s1[r] - mnew);
      float rs = p0 + p1;
      for (int m = 1; m < 16; m <<= 1) rs += __shfl_xor(rs, m, 32);
      lrow[r] = lrow[r] * alpha + rs;
      mrow[r] = mnew;
      o0[r] *= alpha; o1[r] *= alpha; o2[r] *= alpha; o3[r] *= alpha;
      int mr = r + 8 * half;
      pLds[wave][mr * 32 + r16]      = (_Float16)p0;
      pLds[wave][mr * 32 + 16 + r16] = (_Float16)p1;
    }
    asm volatile("s_wait_dscnt 0" ::: "memory");   // in-wave cross-lane LDS RAW

    FragU pf;
    const uint4* sp = (const uint4*)&pLds[wave][0];
    pf.q[0] = sp[r16 * 4 + half];
    pf.q[1] = sp[r16 * 4 + half + 2];

    // ---- O += P * V (V^T fragments from LDS)
    v16h bv0 = load_frag16(sV,      r16, 32, 0, lane);
    v16h bv1 = load_frag16(sV, 16 + r16, 32, 0, lane);
    v16h bv2 = load_frag16(sV, 32 + r16, 32, 0, lane);
    v16h bv3 = load_frag16(sV, 48 + r16, 32, 0, lane);
    o0 = wmma_f16(pf.h, bv0, o0);
    o1 = wmma_f16(pf.h, bv1, o1);
    o2 = wmma_f16(pf.h, bv2, o2);
    o3 = wmma_f16(pf.h, bv3, o3);

    __syncthreads();   // protect sK/sV before next cooperative fill
  }

#pragma unroll
  for (int r = 0; r < 8; ++r) {
    float inv = 1.0f / lrow[r];
    int q = q0 + r + 8 * half;
    size_t base = ((size_t)(b * SEQ + q)) * INNER + h * DHEAD + r16;
    ob[base + 0]  = (_Float16)(o0[r] * inv);
    ob[base + 16] = (_Float16)(o1[r] * inv);
    ob[base + 32] = (_Float16)(o2[r] * inv);
    ob[base + 48] = (_Float16)(o3[r] * inv);
  }
}

// ------- Kernel 5: output projection, 64x32 register-blocked + bias ---------
__global__ void proj_gemm(const _Float16* __restrict__ ao, const _Float16* __restrict__ wot,
                          const float* __restrict__ bout, float* __restrict__ out) {
  int lane = threadIdx.x & 31, wave = threadIdx.x >> 5;
  int t  = blockIdx.x * 4 + wave;
  int mt = t / (DIM / 32);
  int nt = t % (DIM / 32);
  int m0 = mt * 64, n0 = nt * 32;
  int r16 = lane & 15, half = lane >> 4;

  const _Float16* a0 = ao + (size_t)(m0      + r16) * INNER + 8 * half;
  const _Float16* a1 = a0 + 16 * INNER;
  const _Float16* a2 = a0 + 32 * INNER;
  const _Float16* a3 = a0 + 48 * INNER;
  const _Float16* b0 = wot + (size_t)(n0 + r16) * INNER + 8 * half;
  const _Float16* b1 = b0 + 16 * INNER;

  v8f acc[4][2] = {};
  for (int k0 = 0; k0 < INNER; k0 += 32) {
    v16h A0 = load_frag_p(a0, k0);
    v16h A1 = load_frag_p(a1, k0);
    v16h A2 = load_frag_p(a2, k0);
    v16h A3 = load_frag_p(a3, k0);
    v16h B0 = load_frag_p(b0, k0);
    acc[0][0] = wmma_f16(A0, B0, acc[0][0]);
    acc[1][0] = wmma_f16(A1, B0, acc[1][0]);
    acc[2][0] = wmma_f16(A2, B0, acc[2][0]);
    acc[3][0] = wmma_f16(A3, B0, acc[3][0]);
    v16h B1 = load_frag_p(b1, k0);
    acc[0][1] = wmma_f16(A0, B1, acc[0][1]);
    acc[1][1] = wmma_f16(A1, B1, acc[1][1]);
    acc[2][1] = wmma_f16(A2, B1, acc[2][1]);
    acc[3][1] = wmma_f16(A3, B1, acc[3][1]);
  }

#pragma unroll
  for (int j = 0; j < 2; ++j) {
    int col = n0 + 16 * j + r16;
    float bo = bout[col];
#pragma unroll
    for (int i = 0; i < 4; ++i) {
#pragma unroll
      for (int r = 0; r < 8; ++r) {
        int row = m0 + 16 * i + r + 8 * half;
        out[(size_t)row * DIM + col] = acc[i][j][r] + bo;
      }
    }
  }
}

extern "C" void kernel_launch(void* const* d_in, const int* in_sizes, int n_in,
                              void* d_out, int out_size, void* d_ws, size_t ws_size,
                              hipStream_t stream) {
  const float* x     = (const float*)d_in[0];
  const float* g     = (const float*)d_in[1];
  const float* bb    = (const float*)d_in[2];
  const float* w_qkv = (const float*)d_in[3];
  const float* w_out = (const float*)d_in[4];
  const float* b_out = (const float*)d_in[5];

  char* ws = (char*)d_ws;
  _Float16* xn    = (_Float16*)ws;                 ws += (size_t)ROWS * DIM * 2;
  _Float16* wqkvt = (_Float16*)ws;                 ws += (size_t)QKVN * DIM * 2;
  _Float16* wott  = (_Float16*)ws;                 ws += (size_t)DIM * INNER * 2;
  _Float16* qb    = (_Float16*)ws;                 ws += (size_t)ROWS * INNER * 2;
  _Float16* kb    = (_Float16*)ws;                 ws += (size_t)ROWS * INNER * 2;
  _Float16* vtb   = (_Float16*)ws;                 ws += (size_t)ROWS * INNER * 2;
  _Float16* ao    = (_Float16*)ws;                 ws += (size_t)ROWS * INNER * 2;

  ln_kernel<<<ROWS, 256, 0, stream>>>(x, g, bb, xn);
  wtrans_kernel<<<(QKVN * DIM + 255) / 256, 256, 0, stream>>>(w_qkv, wqkvt, DIM, QKVN);
  wtrans_kernel<<<(DIM * INNER + 255) / 256, 256, 0, stream>>>(w_out, wott, INNER, DIM);
  qkv_gemm<<<(ROWS / 64) * (QKVN / 32) / 4, 128, 0, stream>>>(xn, wqkvt, qb, kb, vtb);
  attn_kernel<<<BATCH * HEADS * (SEQ / 64), 128, 0, stream>>>(qb, kb, vtb, ao);
  proj_gemm<<<(ROWS / 64) * (DIM / 32) / 4, 128, 0, stream>>>(ao, wott, b_out, (float*)d_out);
}
